// GroupQueryAttention_84456236908819
// MI455X (gfx1250) — matrix operand order repaired and checked by reading, not compiled, not run on hardware
//
#include <hip/hip_runtime.h>
#include <hip/hip_bf16.h>

typedef _Float16 half_t;
typedef __attribute__((ext_vector_type(16))) _Float16 v16h;
typedef __attribute__((ext_vector_type(8)))  _Float16 h8;
typedef __attribute__((ext_vector_type(8)))  float    v8f;

#define D_IN      2048
#define D_OUT     2048
#define NUM_HEADS 32
#define NUM_KV    8
#define HEAD_DIM  64
#define GRP_SIZE  4
#define BATCH     2
#define SEQ       2048
#define MROWS     (BATCH * SEQ)        // 4096
#define DKV       (NUM_KV * HEAD_DIM)  // 512

union Frag16 { v16h v; h8 h[2]; };

__device__ __forceinline__ v8f zero8() {
    v8f z;
#pragma unroll
    for (int i = 0; i < 8; ++i) z[i] = 0.f;
    return z;
}

// CDNA5 async memory->LDS copy (16B per lane), tracked by ASYNCcnt.
// lds_off = low 32 bits of a generic pointer into __shared__ (LDS aperture:
// addr[31:0] is the LDS offset). gaddr = 64-bit global address.
__device__ __forceinline__ void async_ld_b128(unsigned lds_off, const void* gaddr) {
    asm volatile("global_load_async_to_lds_b128 %0, %1, off"
                 :: "v"(lds_off), "v"(gaddr) : "memory");
}
__device__ __forceinline__ void wait_asynccnt0() {
    asm volatile("s_wait_asynccnt 0" ::: "memory");
}

// ---------------------------------------------------------------------------
// f32 -> f16 conversion
// ---------------------------------------------------------------------------
__global__ void f32_to_f16(const float* __restrict__ src, half_t* __restrict__ dst, size_t n) {
    size_t i = (size_t)blockIdx.x * blockDim.x + threadIdx.x;
    if (i < n) dst[i] = (half_t)src[i];
}

// ---------------------------------------------------------------------------
// RoPE (in place, f16 tensor, f32 cos/sin). Layout: [b][s][head*64 + d]
// ---------------------------------------------------------------------------
__global__ void rope_kernel(half_t* __restrict__ t, const float* __restrict__ cosp,
                            const float* __restrict__ sinp, int nheads) {
    const int HD2 = HEAD_DIM / 2;
    size_t idx = (size_t)blockIdx.x * blockDim.x + threadIdx.x;
    size_t total = (size_t)BATCH * SEQ * nheads * HD2;
    if (idx >= total) return;
    int d = (int)(idx % HD2);
    size_t t2 = idx / HD2;
    int hh = (int)(t2 % nheads);
    size_t t3 = t2 / nheads;
    int s = (int)(t3 % SEQ);
    int bb = (int)(t3 / SEQ);
    half_t* p = t + ((size_t)(bb * SEQ + s) * (size_t)(nheads * HEAD_DIM)) + hh * HEAD_DIM;
    float x1 = (float)p[d];
    float x2 = (float)p[d + HD2];
    float c1 = cosp[s * HEAD_DIM + d],       s1 = sinp[s * HEAD_DIM + d];
    float c2 = cosp[s * HEAD_DIM + d + HD2], s2 = sinp[s * HEAD_DIM + d + HD2];
    p[d]       = (half_t)(x1 * c1 - x2 * s1);
    p[d + HD2] = (half_t)(x2 * c2 + x1 * s2);
}

// ---------------------------------------------------------------------------
// Tiled WMMA GEMM: C[M,N] = A[M,K] (f16, row-major) * B[K,N] (f16, row-major)
// Block = 128x128, BK = 64 (2 WMMA k-steps / barrier), 256 threads (8 waves).
// Wave grid 4(m) x 2(n): each wave owns 32x64 via 2x4 v8f accumulators.
// A tile filled with GLOBAL_LOAD_ASYNC_TO_LDS_B128; B staged transposed.
// ---------------------------------------------------------------------------
template <bool OUTF32>
__global__ __launch_bounds__(256) void gemm_wmma(const half_t* __restrict__ A,
                                                 const half_t* __restrict__ B,
                                                 void* __restrict__ C,
                                                 int M, int N, int K) {
    __shared__ half_t lA[128][72];   // [m][k0..63], pad->72 halfs (144B, conflict-free)
    __shared__ half_t lB[128][72];   // transposed: [n][k]

    const int tid  = threadIdx.x;
    const int wave = tid >> 5;
    const int lane = tid & 31;
    const int lgrp = lane >> 4;      // 0/1 lane half-group
    const int l16  = lane & 15;
    const int wm   = wave & 3;       // m offset wm*32
    const int wn   = wave >> 2;      // n offset wn*64
    const int bM   = blockIdx.y * 128;
    const int bN   = blockIdx.x * 128;

    v8f acc[2][4];
#pragma unroll
    for (int i = 0; i < 2; ++i)
#pragma unroll
        for (int j = 0; j < 4; ++j) acc[i][j] = zero8();

    const int arow = tid >> 1;            // 0..127
    const int acol = (tid & 1) * 32;      // 0 / 32
    const int bk   = tid >> 2;            // 0..63
    const int bn   = (tid & 3) * 32;      // 0,32,64,96

    const unsigned ldsA = (unsigned)(uintptr_t)&lA[arow][acol];

    for (int k0 = 0; k0 < K; k0 += 64) {
        // ---- A tile: 128 x 64 halfs via async copy (4x b128 per thread)
        const half_t* gA = A + (size_t)(bM + arow) * K + k0 + acol;
#pragma unroll
        for (int j = 0; j < 4; ++j)
            async_ld_b128(ldsA + 16u * j, gA + 8 * j);

        // ---- B tile: 64 x 128, transposed into LDS (reg-staged)
        const half_t* gB = B + (size_t)(k0 + bk) * N + bN + bn;
        uint4 br[4];
#pragma unroll
        for (int j = 0; j < 4; ++j) br[j] = ((const uint4*)gB)[j];
        const half_t* hb = (const half_t*)br;
#pragma unroll
        for (int j = 0; j < 32; ++j) lB[bn + j][bk] = hb[j];

        if (k0 + 64 < K) {
            __builtin_prefetch(gA + 64, 0, 1);
            __builtin_prefetch(B + (size_t)(k0 + 64 + bk) * N + bN + bn, 0, 1);
        }
        wait_asynccnt0();
        __syncthreads();

#pragma unroll
        for (int ks = 0; ks < 2; ++ks) {
            // A fragments: lane row = m0 + l16; halfs K = (h%8)+8*lgrp+16*(h/8)
            Frag16 a[2];
#pragma unroll
            for (int mi = 0; mi < 2; ++mi) {
                const half_t* base = &lA[wm * 32 + mi * 16 + l16][ks * 32];
                a[mi].h[0] = *(const h8*)(base + 8 * lgrp);
                a[mi].h[1] = *(const h8*)(base + 16 + 8 * lgrp);
            }
            // B fragments: lane col N = n0 + l16; halfs K = (h%8)+8*lgrp+16*(h/8)
#pragma unroll
            for (int ni = 0; ni < 4; ++ni) {
                Frag16 bf;
                const half_t* base = &lB[wn * 64 + ni * 16 + l16][ks * 32];
                bf.h[0] = *(const h8*)(base + 8 * lgrp);
                bf.h[1] = *(const h8*)(base + 16 + 8 * lgrp);
#pragma unroll
                for (int mi = 0; mi < 2; ++mi) {
                    acc[mi][ni] = __builtin_amdgcn_wmma_f32_16x16x32_f16(
                        false, a[mi].v, false, bf.v, (short)0, acc[mi][ni], false, false);
                }
            }
        }
        __syncthreads();
    }

    // Epilogue: C layout row = r + 8*lgrp, col = l16 within each 16x16 tile
#pragma unroll
    for (int mi = 0; mi < 2; ++mi)
#pragma unroll
        for (int ni = 0; ni < 4; ++ni)
#pragma unroll
            for (int r = 0; r < 8; ++r) {
                const int row = bM + wm * 32 + mi * 16 + r + 8 * lgrp;
                const int col = bN + wn * 64 + ni * 16 + l16;
                if (OUTF32)
                    ((float*)C)[(size_t)row * N + col] = acc[mi][ni][r];
                else
                    ((half_t*)C)[(size_t)row * N + col] = (half_t)acc[mi][ni][r];
            }
}

// ---------------------------------------------------------------------------
// Causal flash attention. Grid: (S/128, NUM_HEADS, BATCH), 256 threads.
// Each wave owns 16 q rows; 64-key tiles (16 WMMAs / iteration) with online
// softmax. K tile filled via async-to-LDS; V staged transposed.
// ---------------------------------------------------------------------------
__global__ __launch_bounds__(256) void fa_kernel(const half_t* __restrict__ Q,
                                                 const half_t* __restrict__ K,
                                                 const half_t* __restrict__ V,
                                                 half_t* __restrict__ ctx) {
    __shared__ half_t lK[64][72];       // [key][d]   (B-frag source for Q*K^T)
    __shared__ half_t lV[64][72];       // [d][key]   (B-frag source for P*V)
    __shared__ half_t lP[8][16][72];    // per-wave P (16 x 64): C->A layout bounce

    const int tid  = threadIdx.x;
    const int wave = tid >> 5;
    const int lane = tid & 31;
    const int lgrp = lane >> 4;
    const int l16  = lane & 15;

    const int qt = blockIdx.x;
    const int h  = blockIdx.y;
    const int b  = blockIdx.z;
    const int g  = h / GRP_SIZE;
    const int qb = qt * 128;
    const int qw = qb + wave * 16;   // wave's first q row

    // Preload this wave's Q fragments (16 rows x 64 d = two K-blocks of 32)
    Frag16 qf[2];
    {
        const half_t* qrow = Q + ((size_t)(b * SEQ + qw + l16) * D_OUT) + h * HEAD_DIM;
#pragma unroll
        for (int kb2 = 0; kb2 < 2; ++kb2) {
            qf[kb2].h[0] = *(const h8*)(qrow + kb2 * 32 + 8 * lgrp);
            qf[kb2].h[1] = *(const h8*)(qrow + kb2 * 32 + 16 + 8 * lgrp);
        }
    }

    float mrow[8], lrow[8];
    v8f o[4];
#pragma unroll
    for (int r = 0; r < 8; ++r) { mrow[r] = -1e30f; lrow[r] = 0.f; }
#pragma unroll
    for (int ni = 0; ni < 4; ++ni) o[ni] = zero8();

    const int nkt  = qt * 2 + 2;        // 64-key tiles up to end of q tile
    const int keyT = tid >> 2;          // 0..63
    const int dB   = (tid & 3) * 16;    // 0,16,32,48

    const unsigned ldsK = (unsigned)(uintptr_t)&lK[keyT][dB];

    for (int kt = 0; kt < nkt; ++kt) {
        const int kb0 = kt * 64;
        const size_t rowoff = ((size_t)(b * SEQ + kb0 + keyT) * DKV) + g * HEAD_DIM + dB;
        // K tile (row-major) via async copy: 2x b128 per thread
        async_ld_b128(ldsK,       K + rowoff);
        async_ld_b128(ldsK + 16u, K + rowoff + 8);
        // V tile transposed (reg-staged scalar DS stores)
        {
            uint4 v0 = *(const uint4*)(V + rowoff);
            uint4 v1 = *(const uint4*)(V + rowoff + 8);
            const half_t* hv0 = (const half_t*)&v0;
            const half_t* hv1 = (const half_t*)&v1;
#pragma unroll
            for (int j = 0; j < 8; ++j) lV[dB + j][keyT]     = hv0[j];
#pragma unroll
            for (int j = 0; j < 8; ++j) lV[dB + 8 + j][keyT] = hv1[j];
        }
        wait_asynccnt0();
        __syncthreads();

        // S = Q * K^T : 4 key subtiles of 16, contraction over d=64 (2 steps)
        v8f sc[4];
#pragma unroll
        for (int nk = 0; nk < 4; ++nk) {
            sc[nk] = zero8();
#pragma unroll
            for (int ks = 0; ks < 2; ++ks) {
                Frag16 bf;
                const half_t* base = &lK[nk * 16 + l16][ks * 32];
                bf.h[0] = *(const h8*)(base + 8 * lgrp);
                bf.h[1] = *(const h8*)(base + 16 + 8 * lgrp);
                sc[nk] = __builtin_amdgcn_wmma_f32_16x16x32_f16(
                    false, qf[ks].v, false, bf.v, (short)0, sc[nk], false, false);
            }
        }

        // Online softmax (rows live in lanes of same half-group; reduce over 16)
#pragma unroll
        for (int r = 0; r < 8; ++r) {
            const int qrow = qw + r + 8 * lgrp;
            float sv[4];
#pragma unroll
            for (int nk = 0; nk < 4; ++nk) {
                sv[nk] = sc[nk][r] * 0.125f;   // 1/sqrt(64)
                if (kb0 + nk * 16 + l16 > qrow) sv[nk] = -1e30f;
            }
            float mt = fmaxf(fmaxf(sv[0], sv[1]), fmaxf(sv[2], sv[3]));
#pragma unroll
            for (int off = 8; off > 0; off >>= 1) mt = fmaxf(mt, __shfl_xor(mt, off, 32));
            const float mo = mrow[r];
            const float mn = fmaxf(mo, mt);
            const float alpha = __expf(mo - mn);
            float rs = 0.f;
#pragma unroll
            for (int nk = 0; nk < 4; ++nk) {
                const float p = __expf(sv[nk] - mn);
                lP[wave][r + 8 * lgrp][nk * 16 + l16] = (half_t)p;
                rs += p;
            }
#pragma unroll
            for (int off = 8; off > 0; off >>= 1) rs += __shfl_xor(rs, off, 32);
            lrow[r] = lrow[r] * alpha + rs;
            mrow[r] = mn;
#pragma unroll
            for (int ni = 0; ni < 4; ++ni) o[ni][r] *= alpha;
        }
        __syncthreads();

        // O += P * V : contraction over 64 keys (2 k-steps)
#pragma unroll
        for (int ks = 0; ks < 2; ++ks) {
            Frag16 pf;
            const half_t* pbase = &lP[wave][l16][ks * 32];
            pf.h[0] = *(const h8*)(pbase + 8 * lgrp);
            pf.h[1] = *(const h8*)(pbase + 16 + 8 * lgrp);
#pragma unroll
            for (int ni = 0; ni < 4; ++ni) {
                Frag16 vf;
                const half_t* vbase = &lV[ni * 16 + l16][ks * 32];
                vf.h[0] = *(const h8*)(vbase + 8 * lgrp);
                vf.h[1] = *(const h8*)(vbase + 16 + 8 * lgrp);
                o[ni] = __builtin_amdgcn_wmma_f32_16x16x32_f16(
                    false, pf.v, false, vf.v, (short)0, o[ni], false, false);
            }
        }
        __syncthreads();
    }

    // Normalize and write ctx (f16), layout [b][s][h*64+d]
#pragma unroll
    for (int r = 0; r < 8; ++r) {
        const int qrow = qw + r + 8 * lgrp;
        const float inv = (lrow[r] > 0.f) ? (1.f / lrow[r]) : 0.f;
        half_t* dst = ctx + ((size_t)(b * SEQ + qrow) * D_OUT) + h * HEAD_DIM + l16;
#pragma unroll
        for (int ni = 0; ni < 4; ++ni) dst[ni * 16] = (half_t)(o[ni][r] * inv);
    }
}

// ---------------------------------------------------------------------------
// Orchestration
// ---------------------------------------------------------------------------
extern "C" void kernel_launch(void* const* d_in, const int* in_sizes, int n_in,
                              void* d_out, int out_size, void* d_ws, size_t ws_size,
                              hipStream_t stream) {
    const float* x    = (const float*)d_in[0];
    const float* cosp = (const float*)d_in[1];
    const float* sinp = (const float*)d_in[2];
    // d_in[3] = bool causal mask: unused (mask computed analytically)
    const float* Wq   = (const float*)d_in[4];
    const float* Wk   = (const float*)d_in[5];
    const float* Wv   = (const float*)d_in[6];
    const float* Wo   = (const float*)d_in[7];

    half_t* ws  = (half_t*)d_ws;
    size_t  off = 0;
    half_t* xh  = ws + off; off += (size_t)MROWS * D_IN;
    half_t* wqh = ws + off; off += (size_t)D_IN * D_OUT;
    half_t* wkh = ws + off; off += (size_t)D_IN * DKV;
    half_t* wvh = ws + off; off += (size_t)D_IN * DKV;
    half_t* woh = ws + off; off += (size_t)D_OUT * D_OUT;
    half_t* qh  = ws + off; off += (size_t)MROWS * D_OUT;
    half_t* kh  = ws + off; off += (size_t)MROWS * DKV;
    half_t* vh  = ws + off; off += (size_t)MROWS * DKV;
    half_t* ch  = ws + off; off += (size_t)MROWS * D_OUT;

    // Stage f16 copies of activations and weights
    {
        size_t n;
        n = (size_t)MROWS * D_IN;  f32_to_f16<<<(n + 255) / 256, 256, 0, stream>>>(x,  xh,  n);
        n = (size_t)D_IN * D_OUT;  f32_to_f16<<<(n + 255) / 256, 256, 0, stream>>>(Wq, wqh, n);
        n = (size_t)D_IN * DKV;    f32_to_f16<<<(n + 255) / 256, 256, 0, stream>>>(Wk, wkh, n);
        n = (size_t)D_IN * DKV;    f32_to_f16<<<(n + 255) / 256, 256, 0, stream>>>(Wv, wvh, n);
        n = (size_t)D_OUT * D_OUT; f32_to_f16<<<(n + 255) / 256, 256, 0, stream>>>(Wo, woh, n);
    }

    // QKV projections
    gemm_wmma<false><<<dim3(D_OUT / 128, MROWS / 128), 256, 0, stream>>>(xh, wqh, qh, MROWS, D_OUT, D_IN);
    gemm_wmma<false><<<dim3(DKV / 128, MROWS / 128),   256, 0, stream>>>(xh, wkh, kh, MROWS, DKV,   D_IN);
    gemm_wmma<false><<<dim3(DKV / 128, MROWS / 128),   256, 0, stream>>>(xh, wvh, vh, MROWS, DKV,   D_IN);

    // RoPE on Q (32 heads) and K (8 kv heads)
    {
        size_t nq = (size_t)BATCH * SEQ * NUM_HEADS * (HEAD_DIM / 2);
        rope_kernel<<<(nq + 255) / 256, 256, 0, stream>>>(qh, cosp, sinp, NUM_HEADS);
        size_t nk = (size_t)BATCH * SEQ * NUM_KV * (HEAD_DIM / 2);
        rope_kernel<<<(nk + 255) / 256, 256, 0, stream>>>(kh, cosp, sinp, NUM_KV);
    }

    // Causal flash attention
    fa_kernel<<<dim3(SEQ / 128, NUM_HEADS, BATCH), 256, 0, stream>>>(qh, kh, vh, ch);

    // Output projection (f32 epilogue straight into d_out)
    gemm_wmma<true><<<dim3(D_OUT / 128, MROWS / 128), 256, 0, stream>>>(ch, woh, d_out, MROWS, D_OUT, D_IN);

    (void)in_sizes; (void)n_in; (void)out_size; (void)ws_size;
}
